// ST_GNN_Identity_53171695124555
// MI455X (gfx1250) — matrix operand order repaired and checked by reading, not compiled, not run on hardware
//
#include <hip/hip_runtime.h>
#include <hip/hip_bf16.h>

typedef __attribute__((ext_vector_type(16))) __bf16 bf16x16;
typedef __attribute__((ext_vector_type(8)))  float  floatx8;

// ---------------------------------------------------------------------------
// helpers
// ---------------------------------------------------------------------------
__device__ __forceinline__ unsigned f2ord(float f) {
    unsigned b = __float_as_uint(f);
    return (b & 0x80000000u) ? ~b : (b | 0x80000000u);
}
__device__ __forceinline__ float ord2f(unsigned u) {
    unsigned b = (u & 0x80000000u) ? (u ^ 0x80000000u) : ~u;
    return __uint_as_float(b);
}

__global__ void fill_f32(float* p, float v, size_t n) {
    size_t t = (size_t)blockIdx.x * blockDim.x + threadIdx.x;
    if (t < n) p[t] = v;
}
__global__ void fill_u32(unsigned* p, unsigned v, size_t n) {
    size_t t = (size_t)blockIdx.x * blockDim.x + threadIdx.x;
    if (t < n) p[t] = v;
}

// ---------------------------------------------------------------------------
// Generic bf16 WMMA GEMM: C[M,Nn] = act(A[M,K] * B[K,Nn] + bias)
// One wave per 16x16 output tile, K stepped by 32.
// Requires: M%16==0, K%32==0, Nn%16==0 (true for every call site here).
// act: 0 = none, 1 = relu
// ---------------------------------------------------------------------------
__global__ void __launch_bounds__(128)
gemm_bias_act(const float* __restrict__ A, const float* __restrict__ B,
              const float* __restrict__ bias, float* __restrict__ C,
              int M, int K, int Nn, int act)
{
    const int wave = threadIdx.x >> 5;
    const int lane = threadIdx.x & 31;
    const int tilesN = Nn >> 4;
    const int tiles  = (M >> 4) * tilesN;
    const int tid = blockIdx.x * 4 + wave;        // wave-uniform
    if (tid >= tiles) return;                     // whole wave exits together
    const int tm = tid / tilesN;
    const int tn = tid - tm * tilesN;
    const int half = lane >> 4;
    const int l16  = lane & 15;

    const float* Arow = A + (size_t)(tm * 16 + l16) * K;  // A-fragment row
    const int colg = tn * 16 + l16;                       // B/C column

    floatx8 c;
#pragma unroll
    for (int r = 0; r < 8; r++) c[r] = 0.0f;

    for (int kk = 0; kk < K; kk += 32) {
        if (kk + 32 < K) {
            // CDNA5 global_prefetch_b8 for the next K-step (A row + B panel)
            __builtin_prefetch(Arow + kk + 32, 0, 1);
            __builtin_prefetch(B + (size_t)(kk + 32 + 16 * half) * Nn + colg, 0, 1);
        }
        bf16x16 a, b;
#pragma unroll
        for (int i = 0; i < 8; i++) {
            // A 16x32 bf16 layout: VGPR i<4 -> K = 8*half + 2i ; i>=4 -> 16 + 8*half + 2(i-4)
            int k0 = kk + ((i < 4) ? (8 * half + 2 * i) : (16 + 8 * half + 2 * (i - 4)));
            float2 av = *(const float2*)(Arow + k0);       // contiguous pair -> b64 load
            a[2 * i]     = (__bf16)av.x;
            a[2 * i + 1] = (__bf16)av.y;
            // B 32x16 bf16 layout: VGPR i<4 -> K = 8*half + 2i ; i>=4 -> 16 + 8*half + 2(i-4)
            int kb = kk + ((i < 4) ? (8 * half + 2 * i) : (16 + 8 * half + 2 * (i - 4)));
            b[2 * i]     = (__bf16)B[(size_t)kb * Nn + colg];
            b[2 * i + 1] = (__bf16)B[(size_t)(kb + 1) * Nn + colg];
        }
        c = __builtin_amdgcn_wmma_f32_16x16x32_bf16(false, a, false, b, (short)0, c,
                                                    false, false);
    }

    const float bv = bias ? bias[colg] : 0.0f;
#pragma unroll
    for (int r = 0; r < 8; r++) {
        int m = tm * 16 + r + 8 * half;           // C/D layout: VGPR r -> M = r + 8*half
        float v = c[r] + bv;
        if (act == 1) v = v > 0.0f ? v : 0.0f;
        C[(size_t)m * Nn + colg] = v;
    }
}

// ---------------------------------------------------------------------------
// Stage A: edge-feature scatter-mean into node input features
// ---------------------------------------------------------------------------
__global__ void scatter_ee(const float* __restrict__ ee, const int* __restrict__ src,
                           const int* __restrict__ dst, float* ssum, float* dsum,
                           long long total)
{
    long long t = (long long)blockIdx.x * blockDim.x + threadIdx.x;
    if (t >= total) return;
    int e = (int)(t >> 6), f = (int)(t & 63);
    float v = ee[t];
    atomicAdd(&ssum[(size_t)src[e] * 64 + f], v);
    atomicAdd(&dsum[(size_t)dst[e] * 64 + f], v);
}

__global__ void count_edges(const int* __restrict__ src, const int* __restrict__ dst,
                            float* cs, float* cd, int E)
{
    int e = blockIdx.x * blockDim.x + threadIdx.x;
    if (e >= E) return;
    atomicAdd(&cs[src[e]], 1.0f);
    atomicAdd(&cd[dst[e]], 1.0f);
}

__global__ void finalize_xin(const float* ssum, const float* dsum, const float* cs,
                             const float* cd, float* xin, int total /* N*64 */)
{
    int t = blockIdx.x * blockDim.x + threadIdx.x;
    if (t >= total) return;
    int n = t >> 6, f = t & 63;
    xin[(size_t)n * 128 + f]      = ssum[t] / fmaxf(cs[n], 1.0f);
    xin[(size_t)n * 128 + 64 + f] = dsum[t] / fmaxf(cd[n], 1.0f);
}

// ---------------------------------------------------------------------------
// GATv2 per-edge attention logits (edge_attr @ We fused, We cached in LDS)
// logits[e,h] = sum_d att[h,d] * leaky_relu(xl[src,h,d] + xr[dst,h,d] + ee1[e,h,d], 0.2)
// ---------------------------------------------------------------------------
__global__ void gat_edge_logits(const float* __restrict__ edge_attr,
                                const int* __restrict__ src, const int* __restrict__ dst,
                                const float* __restrict__ xl, const float* __restrict__ xr,
                                const float* __restrict__ We, const float* __restrict__ att,
                                float* __restrict__ logits, int E, int H)
{
    extern __shared__ float sWe[];                 // [32 x H*128]
    const int ld = H * 128;
    for (int i = threadIdx.x; i < 32 * ld; i += blockDim.x) sWe[i] = We[i];
    __syncthreads();

    int t = blockIdx.x * blockDim.x + threadIdx.x; // t = e*H + h
    if (t >= E * H) return;
    int e = t / H, h = t - e * H;

    const float* xls = xl + (size_t)src[e] * ld + h * 128;
    const float* xrd = xr + (size_t)dst[e] * ld + h * 128;
    __builtin_prefetch(xls, 0, 1);                 // random gathered rows
    __builtin_prefetch(xrd, 0, 1);

    float ea[32];
#pragma unroll
    for (int k = 0; k < 32; k++) ea[k] = edge_attr[(size_t)e * 32 + k];

    const float* at  = att + h * 128;
    const float* we  = sWe + h * 128;

    float acc = 0.0f;
    for (int d = 0; d < 128; d++) {
        float eev = 0.0f;
#pragma unroll
        for (int k = 0; k < 32; k++) eev += ea[k] * we[k * ld + d];
        float m = xls[d] + xrd[d] + eev;
        m = m > 0.0f ? m : 0.2f * m;               // leaky_relu(0.2)
        acc += at[d] * m;
    }
    logits[t] = acc;
}

__global__ void seg_max(const float* __restrict__ logits, const int* __restrict__ dst,
                        unsigned* mxu, int total, int H)
{
    int t = blockIdx.x * blockDim.x + threadIdx.x;
    if (t >= total) return;
    int e = t / H, h = t - e * H;
    atomicMax(&mxu[(size_t)dst[e] * H + h], f2ord(logits[t]));
}

__global__ void seg_exp(float* __restrict__ logits, const int* __restrict__ dst,
                        const unsigned* __restrict__ mxu, float* den, int total, int H)
{
    int t = blockIdx.x * blockDim.x + threadIdx.x;
    if (t >= total) return;
    int e = t / H, h = t - e * H;
    float m  = ord2f(mxu[(size_t)dst[e] * H + h]);
    float ex = expf(logits[t] - m);
    logits[t] = ex;                                 // reuse buffer for exp values
    atomicAdd(&den[(size_t)dst[e] * H + h], ex);
}

__global__ void gat_agg(const float* __restrict__ ex, const int* __restrict__ src,
                        const int* __restrict__ dst, const float* __restrict__ den,
                        const float* __restrict__ xl, float* agg, long long total, int H)
{
    long long t = (long long)blockIdx.x * blockDim.x + threadIdx.x;
    if (t >= total) return;
    int e = (int)(t >> 7), d = (int)(t & 127);
    int s = src[e], dn = dst[e];
    const int ld = H * 128;
    for (int h = 0; h < H; h++) {
        float alpha = ex[(size_t)e * H + h] / (den[(size_t)dn * H + h] + 1e-16f);
        atomicAdd(&agg[(size_t)dn * ld + h * 128 + d],
                  alpha * xl[(size_t)s * ld + h * 128 + d]);
    }
}

__global__ void head_mean_bias(const float* __restrict__ agg, const float* __restrict__ bias,
                               float* z, int total /* N*128 */, int H)
{
    int t = blockIdx.x * blockDim.x + threadIdx.x;
    if (t >= total) return;
    int n = t >> 7, d = t & 127;
    const int ld = H * 128;
    float s = 0.0f;
    for (int h = 0; h < H; h++) s += agg[(size_t)n * ld + h * 128 + d];
    z[t] = s / (float)H + bias[d];
}

__global__ void ln_elu(const float* __restrict__ in, const float* __restrict__ g,
                       const float* __restrict__ b, float* out, int N)
{
    int n = blockIdx.x * blockDim.x + threadIdx.x;
    if (n >= N) return;
    const float* r = in + (size_t)n * 128;
    float mu = 0.0f;
    for (int i = 0; i < 128; i++) mu += r[i];
    mu *= (1.0f / 128.0f);
    float var = 0.0f;
    for (int i = 0; i < 128; i++) { float d = r[i] - mu; var += d * d; }
    var *= (1.0f / 128.0f);
    float inv = rsqrtf(var + 1e-5f);
    float* o = out + (size_t)n * 128;
    for (int i = 0; i < 128; i++) {
        float v = (r[i] - mu) * inv * g[i] + b[i];
        o[i] = v > 0.0f ? v : expm1f(v);           // elu
    }
}

// torch GRUCell, gate order (r, z, n)
__global__ void gru_ew(const float* __restrict__ gi, const float* __restrict__ gh,
                       const float* __restrict__ hprev, float* hcur, int total /* N*128 */)
{
    int t = blockIdx.x * blockDim.x + threadIdx.x;
    if (t >= total) return;
    int n = t >> 7, d = t & 127;
    const float* gin = gi + (size_t)n * 384;
    const float* ghn = gh + (size_t)n * 384;
    float r = 1.0f / (1.0f + expf(-(gin[d] + ghn[d])));
    float u = 1.0f / (1.0f + expf(-(gin[128 + d] + ghn[128 + d])));
    float cand = tanhf(gin[256 + d] + r * ghn[256 + d]);
    float hp = hprev[t];
    hcur[t] = (1.0f - u) * cand + u * hp;
}

// ---------------------------------------------------------------------------
// Fused edge classifier: rep[E,288] -> relu GEMM 288x128 -> relu GEMM 128x64 -> dot 64x1
// One wave per 16 edges; A-fragments gathered straight from h_cur/edge_attr.
// ---------------------------------------------------------------------------
__device__ __forceinline__ float rep_val(const float* __restrict__ h, int s, int d,
                                         const float* __restrict__ ea, int e, int k)
{
    if (k < 128) return h[(size_t)s * 128 + k];
    if (k < 256) return h[(size_t)d * 128 + (k - 128)];
    return ea[(size_t)e * 32 + (k - 256)];
}

__global__ void __launch_bounds__(128)
cls_kernel(const float* __restrict__ hcur, const int* __restrict__ src,
           const int* __restrict__ dst, const float* __restrict__ edge_attr,
           const float* __restrict__ W1, const float* __restrict__ b1,
           const float* __restrict__ W2, const float* __restrict__ b2,
           const float* __restrict__ W3, const float* __restrict__ b3,
           float* __restrict__ out, int E)
{
    __shared__ float h1buf[4][16][128];            // 32 KB
    __shared__ float h2buf[4][16][64];             // 16 KB
    const int wave = threadIdx.x >> 5;
    const int lane = threadIdx.x & 31;
    const int grp = blockIdx.x * 4 + wave;         // wave-uniform
    if (grp >= (E >> 4)) return;
    const int e0 = grp * 16;
    const int half = lane >> 4;
    const int col  = lane & 15;
    const int arow = lane & 15;
    const int eA = e0 + arow;
    const int sA = src[eA], dA = dst[eA];

    // prefetch the randomly-gathered node rows for this wave's 16 edges
    __builtin_prefetch(hcur + (size_t)sA * 128, 0, 1);
    __builtin_prefetch(hcur + (size_t)dA * 128, 0, 1);

    // ---- layer 1: 16x288 * 288x128 ----
    for (int nt = 0; nt < 8; nt++) {
        floatx8 c;
#pragma unroll
        for (int r = 0; r < 8; r++) c[r] = 0.0f;
        for (int kk = 0; kk < 9; kk++) {           // K = 288
            bf16x16 a, b;
#pragma unroll
            for (int i = 0; i < 8; i++) {
                int k0 = kk * 32 + ((i < 4) ? (8 * half + 2 * i)
                                            : (16 + 8 * half + 2 * (i - 4)));
                a[2 * i]     = (__bf16)rep_val(hcur, sA, dA, edge_attr, eA, k0);
                a[2 * i + 1] = (__bf16)rep_val(hcur, sA, dA, edge_attr, eA, k0 + 1);
                int kb = k0;
                b[2 * i]     = (__bf16)W1[(size_t)kb * 128 + nt * 16 + col];
                b[2 * i + 1] = (__bf16)W1[(size_t)(kb + 1) * 128 + nt * 16 + col];
            }
            c = __builtin_amdgcn_wmma_f32_16x16x32_bf16(false, a, false, b, (short)0, c,
                                                        false, false);
        }
#pragma unroll
        for (int r = 0; r < 8; r++) {
            int m = r + 8 * half;
            float v = c[r] + b1[nt * 16 + col];
            h1buf[wave][m][nt * 16 + col] = v > 0.0f ? v : 0.0f;
        }
    }

    // ---- layer 2: 16x128 * 128x64 ----  (same-wave LDS ops are in-order)
    for (int nt = 0; nt < 4; nt++) {
        floatx8 c;
#pragma unroll
        for (int r = 0; r < 8; r++) c[r] = 0.0f;
        for (int kk = 0; kk < 4; kk++) {           // K = 128
            bf16x16 a, b;
#pragma unroll
            for (int i = 0; i < 8; i++) {
                int k0 = kk * 32 + ((i < 4) ? (8 * half + 2 * i)
                                            : (16 + 8 * half + 2 * (i - 4)));
                float2 av = *(const float2*)(&h1buf[wave][arow][k0]);  // ds b64
                a[2 * i]     = (__bf16)av.x;
                a[2 * i + 1] = (__bf16)av.y;
                int kb = k0;
                b[2 * i]     = (__bf16)W2[(size_t)kb * 64 + nt * 16 + col];
                b[2 * i + 1] = (__bf16)W2[(size_t)(kb + 1) * 64 + nt * 16 + col];
            }
            c = __builtin_amdgcn_wmma_f32_16x16x32_bf16(false, a, false, b, (short)0, c,
                                                        false, false);
        }
#pragma unroll
        for (int r = 0; r < 8; r++) {
            int m = r + 8 * half;
            float v = c[r] + b2[nt * 16 + col];
            h2buf[wave][m][nt * 16 + col] = v > 0.0f ? v : 0.0f;
        }
    }

    // ---- layer 3: 64 -> 1 dot, lanes 0..15 each own one edge ----
    if (lane < 16) {
        float s = b3[0];
        for (int j = 0; j < 64; j++) s += h2buf[wave][lane][j] * W3[j];
        out[e0 + lane] = s;
    }
}

// ---------------------------------------------------------------------------
// Orchestration
// ---------------------------------------------------------------------------
static inline unsigned cdiv_u(size_t a, size_t b) { return (unsigned)((a + b - 1) / b); }

extern "C" void kernel_launch(void* const* d_in, const int* in_sizes, int n_in,
                              void* d_out, int out_size, void* d_ws, size_t ws_size,
                              hipStream_t stream)
{
    (void)n_in; (void)out_size; (void)ws_size;
    const int*   edge_index = (const int*)d_in[1];
    const float* edge_attr  = (const float*)d_in[2];
    const float* h_prev     = (const float*)d_in[3];
    const float* eW   = (const float*)d_in[4],  *eb   = (const float*)d_in[5];
    const float* g1Wl = (const float*)d_in[6],  *g1bl = (const float*)d_in[7];
    const float* g1Wr = (const float*)d_in[8],  *g1br = (const float*)d_in[9];
    const float* g1We = (const float*)d_in[10], *g1at = (const float*)d_in[11];
    const float* g1bs = (const float*)d_in[12];
    const float* ln1g = (const float*)d_in[13], *ln1b = (const float*)d_in[14];
    const float* g2Wl = (const float*)d_in[15], *g2bl = (const float*)d_in[16];
    const float* g2Wr = (const float*)d_in[17], *g2br = (const float*)d_in[18];
    const float* g2We = (const float*)d_in[19], *g2at = (const float*)d_in[20];
    const float* g2bs = (const float*)d_in[21];
    const float* ln2g = (const float*)d_in[22], *ln2b = (const float*)d_in[23];
    const float* gWi  = (const float*)d_in[24], *gWh  = (const float*)d_in[25];
    const float* gbi  = (const float*)d_in[26], *gbh  = (const float*)d_in[27];
    const float* cW1  = (const float*)d_in[28], *cb1  = (const float*)d_in[29];
    const float* cW2  = (const float*)d_in[30], *cb2  = (const float*)d_in[31];
    const float* cW3  = (const float*)d_in[32], *cb3  = (const float*)d_in[33];

    const int E  = in_sizes[1] / 2;      // edge_index is [2,E]
    const int NV = in_sizes[3] / 128;    // h_prev is [N,128]
    const int* src = edge_index;
    const int* dst = edge_index + E;

    size_t off = 0;
    auto alloc = [&](size_t elems) -> float* {
        float* p = (float*)((char*)d_ws + off);
        off += (elems * sizeof(float) + 255) & ~(size_t)255;
        return p;
    };
    float*    ee   = alloc((size_t)E * 64);
    float*    ssum = alloc((size_t)NV * 64);
    float*    dsum = alloc((size_t)NV * 64);
    float*    cs   = alloc((size_t)NV);
    float*    cd   = alloc((size_t)NV);
    float*    xin  = alloc((size_t)NV * 128);
    float*    xl   = alloc((size_t)NV * 256);
    float*    xr   = alloc((size_t)NV * 256);
    float*    lg   = alloc((size_t)E * 2);
    unsigned* mxu  = (unsigned*)alloc((size_t)NV * 2);
    float*    den  = alloc((size_t)NV * 2);
    float*    agg  = alloc((size_t)NV * 256);
    float*    z1   = alloc((size_t)NV * 128);
    float*    z2   = alloc((size_t)NV * 128);
    float*    gi   = alloc((size_t)NV * 384);
    float*    gh   = alloc((size_t)NV * 384);
    float*    hcur = alloc((size_t)NV * 128);

    const unsigned NEG_INF_ORD = 0x007FFFFFu;      // f2ord(-inf)

    // ---- Stage A: ee = relu(edge_attr @ edge_proj_W + b); scatter-mean -> x_input
    gemm_bias_act<<<cdiv_u((size_t)(E / 16) * 4, 4), 128, 0, stream>>>(
        edge_attr, eW, eb, ee, E, 32, 64, 1);
    fill_f32<<<cdiv_u((size_t)NV * 64, 256), 256, 0, stream>>>(ssum, 0.0f, (size_t)NV * 64);
    fill_f32<<<cdiv_u((size_t)NV * 64, 256), 256, 0, stream>>>(dsum, 0.0f, (size_t)NV * 64);
    fill_f32<<<cdiv_u(NV, 256), 256, 0, stream>>>(cs, 0.0f, NV);
    fill_f32<<<cdiv_u(NV, 256), 256, 0, stream>>>(cd, 0.0f, NV);
    scatter_ee<<<cdiv_u((size_t)E * 64, 256), 256, 0, stream>>>(ee, src, dst, ssum, dsum,
                                                               (long long)E * 64);
    count_edges<<<cdiv_u(E, 256), 256, 0, stream>>>(src, dst, cs, cd, E);
    finalize_xin<<<cdiv_u((size_t)NV * 64, 256), 256, 0, stream>>>(ssum, dsum, cs, cd, xin,
                                                                   NV * 64);

    // ---- GAT layer 1 (H=2, out=128)
    gemm_bias_act<<<cdiv_u((size_t)(NV / 16) * 16, 4), 128, 0, stream>>>(
        xin, g1Wl, g1bl, xl, NV, 128, 256, 0);
    gemm_bias_act<<<cdiv_u((size_t)(NV / 16) * 16, 4), 128, 0, stream>>>(
        xin, g1Wr, g1br, xr, NV, 128, 256, 0);
    fill_u32<<<cdiv_u((size_t)NV * 2, 256), 256, 0, stream>>>(mxu, NEG_INF_ORD, (size_t)NV * 2);
    fill_f32<<<cdiv_u((size_t)NV * 2, 256), 256, 0, stream>>>(den, 0.0f, (size_t)NV * 2);
    fill_f32<<<cdiv_u((size_t)NV * 256, 256), 256, 0, stream>>>(agg, 0.0f, (size_t)NV * 256);
    gat_edge_logits<<<cdiv_u((size_t)E * 2, 256), 256, 32 * 256 * 4, stream>>>(
        edge_attr, src, dst, xl, xr, g1We, g1at, lg, E, 2);
    seg_max<<<cdiv_u((size_t)E * 2, 256), 256, 0, stream>>>(lg, dst, mxu, E * 2, 2);
    seg_exp<<<cdiv_u((size_t)E * 2, 256), 256, 0, stream>>>(lg, dst, mxu, den, E * 2, 2);
    gat_agg<<<cdiv_u((size_t)E * 128, 256), 256, 0, stream>>>(lg, src, dst, den, xl, agg,
                                                              (long long)E * 128, 2);
    head_mean_bias<<<cdiv_u((size_t)NV * 128, 256), 256, 0, stream>>>(agg, g1bs, z1,
                                                                      NV * 128, 2);
    ln_elu<<<cdiv_u(NV, 128), 128, 0, stream>>>(z1, ln1g, ln1b, z1, NV);

    // ---- GAT layer 2 (H=1, out=128)
    gemm_bias_act<<<cdiv_u((size_t)(NV / 16) * 8, 4), 128, 0, stream>>>(
        z1, g2Wl, g2bl, xl, NV, 128, 128, 0);
    gemm_bias_act<<<cdiv_u((size_t)(NV / 16) * 8, 4), 128, 0, stream>>>(
        z1, g2Wr, g2br, xr, NV, 128, 128, 0);
    fill_u32<<<cdiv_u(NV, 256), 256, 0, stream>>>(mxu, NEG_INF_ORD, NV);
    fill_f32<<<cdiv_u(NV, 256), 256, 0, stream>>>(den, 0.0f, NV);
    fill_f32<<<cdiv_u((size_t)NV * 128, 256), 256, 0, stream>>>(agg, 0.0f, (size_t)NV * 128);
    gat_edge_logits<<<cdiv_u(E, 256), 256, 32 * 128 * 4, stream>>>(
        edge_attr, src, dst, xl, xr, g2We, g2at, lg, E, 1);
    seg_max<<<cdiv_u(E, 256), 256, 0, stream>>>(lg, dst, mxu, E, 1);
    seg_exp<<<cdiv_u(E, 256), 256, 0, stream>>>(lg, dst, mxu, den, E, 1);
    gat_agg<<<cdiv_u((size_t)E * 128, 256), 256, 0, stream>>>(lg, src, dst, den, xl, agg,
                                                              (long long)E * 128, 1);
    head_mean_bias<<<cdiv_u((size_t)NV * 128, 256), 256, 0, stream>>>(agg, g2bs, z2,
                                                                      NV * 128, 1);
    ln_elu<<<cdiv_u(NV, 128), 128, 0, stream>>>(z2, ln2g, ln2b, z2, NV);

    // ---- GRU cell
    gemm_bias_act<<<cdiv_u((size_t)(NV / 16) * 24, 4), 128, 0, stream>>>(
        z2, gWi, gbi, gi, NV, 128, 384, 0);
    gemm_bias_act<<<cdiv_u((size_t)(NV / 16) * 24, 4), 128, 0, stream>>>(
        h_prev, gWh, gbh, gh, NV, 128, 384, 0);
    gru_ew<<<cdiv_u((size_t)NV * 128, 256), 256, 0, stream>>>(gi, gh, h_prev, hcur, NV * 128);

    // ---- fused edge classifier -> d_out [E]
    cls_kernel<<<cdiv_u((size_t)(E / 16), 4), 128, 0, stream>>>(
        hcur, src, dst, edge_attr, cW1, cb1, cW2, cb2, cW3, cb3, (float*)d_out, E);
}
